// Faster_RCNN_27358941675977
// MI455X (gfx1250) — matrix operand order repaired and checked by reading, not compiled, not run on hardware
//
#include <hip/hip_runtime.h>
#include <hip/hip_bf16.h>
#include <math.h>

typedef __attribute__((ext_vector_type(16))) _Float16 v16h;
typedef __attribute__((ext_vector_type(8)))  float    v8f;
typedef __attribute__((ext_vector_type(8)))  unsigned v8u;

#define FHW 96
#define CCH 512
#define NA 9
#define NANCH (FHW*FHW*NA)      // 82944
#define MROWS (FHW*FHW)         // 9216
#define K9 (9*CCH)              // 4608
#define MAXOUT 1000
#define NMS_T 0.1f

// LDS byte offset of a __shared__ pointer: generic LDS addresses carry the
// wave-relative LDS byte address in bits [31:0] (ISA 10.2 aperture mapping).
#define LDS_OFF(p) ((unsigned)(size_t)(p))

// ---------------------------------------------------------------- converts
__global__ __launch_bounds__(256) void k_cvt_f16(const float* __restrict__ src,
                                                 _Float16* __restrict__ dst, int n) {
    int i = blockIdx.x * 256 + threadIdx.x;
    if (i < n) dst[i] = (_Float16)src[i];
}

// pad feature [96,96,512] -> [98,98,512] f16, zero border
__global__ __launch_bounds__(256) void k_pad_feat(const float* __restrict__ f,
                                                  _Float16* __restrict__ p) {
    int idx = blockIdx.x * 256 + threadIdx.x;
    if (idx >= 98 * 98 * 512) return;
    int c = idx & 511;
    int r = idx >> 9;
    int x = r % 98, y = r / 98;
    _Float16 v = (_Float16)0.0f;
    if (x >= 1 && x <= 96 && y >= 1 && y <= 96)
        v = (_Float16)f[((size_t)(y - 1) * 96 + (x - 1)) * 512 + c];
    p[idx] = v;
}

// pack W_cls [512,18] + W_box [512,36] -> [512,64] f16 (cols 54..63 zero)
__global__ __launch_bounds__(256) void k_pack_wcb(const float* __restrict__ wc,
                                                  const float* __restrict__ wb,
                                                  _Float16* __restrict__ o) {
    int idx = blockIdx.x * 256 + threadIdx.x;
    if (idx >= 512 * 64) return;
    int cin = idx >> 6, j = idx & 63;
    float v = 0.f;
    if (j < 18)      v = wc[cin * 18 + j];
    else if (j < 54) v = wb[cin * 36 + (j - 18)];
    o[idx] = (_Float16)v;
}

// ---------------------------------------------------------------- 3x3 conv as WMMA GEMM
// C[9216,512] = im2col(padF)[9216,4608] x W16[4608,512], +bias, ReLU, f16 out
// 256 threads = 8 waves; block tile 128(M) x 64(N); wave tile 32x32 (2x2 WMMA)
// Software pipeline: async A tile -> LDS (ASYNCcnt), double-buffered, 1 barrier/step.
__global__ __launch_bounds__(256) void k_conv_gemm(const _Float16* __restrict__ padF,
                                                   const _Float16* __restrict__ W16,
                                                   const float* __restrict__ brpn,
                                                   _Float16* __restrict__ X16) {
    __shared__ _Float16 lA[2][128 * 40];   // [row][k], stride 40 halfs (bank-conflict-free)
    __shared__ _Float16 lB[2][64 * 40];    // [col][k] transposed

    const int tid  = threadIdx.x;
    const int lane = tid & 31;
    const int wave = tid >> 5;
    const int wm = wave >> 1, wn = wave & 1;
    const int m0 = blockIdx.y * 128, n0 = blockIdx.x * 64;

    v8f acc[2][2] = {};

    const int arow = tid >> 1, aoff = (tid & 1) * 16;   // A loader: 32B each (2x b128)
    const int bkrow = tid >> 3, bnc0 = (tid & 7) * 8;   // B loader: 8 halfs each
    const int am = m0 + arow;
    const int ay = am / 96, ax = am % 96;

    auto a_src = [&](int s) -> const _Float16* {
        const int kk0 = s * 32;
        const int f = kk0 >> 9, cin0 = kk0 & 511;
        return padF + ((size_t)(ay + f / 3) * 98 + (ax + f % 3)) * 512 + cin0 + aoff;
    };
    auto stageA = [&](int s, int buf) {   // async global -> LDS, no VGPR round trip
        unsigned long long src = (unsigned long long)(size_t)a_src(s);
        unsigned dst = LDS_OFF(&lA[buf][arow * 40 + aoff]);
        asm volatile("global_load_async_to_lds_b128 %0, %1, off"
                     :: "v"(dst), "v"(src) : "memory");
        asm volatile("global_load_async_to_lds_b128 %0, %1, off offset:16"
                     :: "v"(dst), "v"(src) : "memory");  // INST_OFFSET shifts LDS+global
    };
    auto stageB = [&](int s, int buf) {   // transpose scatter -> per-lane LDS, VGPR path
        const _Float16* bs = W16 + (size_t)(s * 32 + bkrow) * 512 + n0 + bnc0;
        _Float16 tmp[8];
        *(uint4*)tmp = *(const uint4*)bs;
#pragma unroll
        for (int j = 0; j < 8; ++j) lB[buf][(bnc0 + j) * 40 + bkrow] = tmp[j];
    };

    stageA(0, 0);
    stageB(0, 0);
    asm volatile("s_wait_asynccnt 0x0" ::: "memory");
    __syncthreads();

    for (int s = 0; s < 144; ++s) {
        const int cur = s & 1;
        if (s + 1 < 144) {                      // fill other buffer while computing
            stageA(s + 1, cur ^ 1);
            stageB(s + 1, cur ^ 1);
        }
        if (s + 2 < 144) {                      // feed L2 two steps ahead
            __builtin_prefetch(a_src(s + 2), 0, 1);
            __builtin_prefetch(W16 + (size_t)((s + 2) * 32 + bkrow) * 512 + n0 + bnc0, 0, 1);
        }

        const int kh = (lane >> 4) * 8;         // A frag: upper half-wave -> K+8
        const int kb = (lane >> 4) * 8;         // B frag: upper half-wave -> K+8
#pragma unroll
        for (int mi = 0; mi < 2; ++mi) {
            const int rl = wm * 32 + mi * 16 + (lane & 15);
            v8u au;
#pragma unroll
            for (int v = 0; v < 8; ++v) {
                int kpos = ((v & 3) << 1) + ((v >= 4) ? 16 : 0) + kh;
                au[v] = *(const unsigned*)(&lA[cur][rl * 40 + kpos]);
            }
            v16h afr = __builtin_bit_cast(v16h, au);
#pragma unroll
            for (int ni = 0; ni < 2; ++ni) {
                const int ncol = wn * 32 + ni * 16 + (lane & 15);
                v8u bu;
#pragma unroll
                for (int v = 0; v < 8; ++v)
                    bu[v] = *(const unsigned*)(&lB[cur][ncol * 40 + kb + ((v & 3) << 1) + ((v >= 4) ? 16 : 0)]);
                v16h bfr = __builtin_bit_cast(v16h, bu);
                acc[mi][ni] = __builtin_amdgcn_wmma_f32_16x16x32_f16(
                    false, afr, false, bfr, (short)0, acc[mi][ni], false, false);
            }
        }

        if (s + 1 < 144) {
            asm volatile("s_wait_asynccnt 0x0" ::: "memory");  // next tile landed in LDS
            __syncthreads();                                   // single barrier per step
        }
    }

    // epilogue: bias + ReLU + f16 store (C layout: VGPR r -> M=r, +8 for upper lanes)
#pragma unroll
    for (int mi = 0; mi < 2; ++mi)
#pragma unroll
        for (int ni = 0; ni < 2; ++ni) {
            const int nc = n0 + wn * 32 + ni * 16 + (lane & 15);
            const float bias = brpn[nc];
            const int mbase = m0 + wm * 32 + mi * 16 + ((lane >> 4) << 3);
#pragma unroll
            for (int r = 0; r < 8; ++r) {
                float v = acc[mi][ni][r] + bias;
                v = v > 0.f ? v : 0.f;
                X16[(size_t)(mbase + r) * 512 + nc] = (_Float16)v;
            }
        }
}

// ---------------------------------------------------------------- 1x1 heads as WMMA GEMM
// head[9216,64] = X16[9216,512] x Wcb[512,64] + bias  (f32 out), same pipeline
__global__ __launch_bounds__(256) void k_head_gemm(const _Float16* __restrict__ X16,
                                                   const _Float16* __restrict__ Wcb,
                                                   const float* __restrict__ bcls,
                                                   const float* __restrict__ bbx,
                                                   float* __restrict__ head) {
    __shared__ _Float16 lA[2][128 * 40];
    __shared__ _Float16 lB[2][64 * 40];

    const int tid  = threadIdx.x;
    const int lane = tid & 31;
    const int wave = tid >> 5;
    const int wm = wave >> 1, wn = wave & 1;
    const int m0 = blockIdx.y * 128;

    v8f acc[2][2] = {};

    const int arow = tid >> 1, aoff = (tid & 1) * 16;
    const int bkrow = tid >> 3, bnc0 = (tid & 7) * 8;

    auto stageA = [&](int s, int buf) {
        unsigned long long src =
            (unsigned long long)(size_t)(X16 + (size_t)(m0 + arow) * 512 + s * 32 + aoff);
        unsigned dst = LDS_OFF(&lA[buf][arow * 40 + aoff]);
        asm volatile("global_load_async_to_lds_b128 %0, %1, off"
                     :: "v"(dst), "v"(src) : "memory");
        asm volatile("global_load_async_to_lds_b128 %0, %1, off offset:16"
                     :: "v"(dst), "v"(src) : "memory");
    };
    auto stageB = [&](int s, int buf) {
        const _Float16* bs = Wcb + (size_t)(s * 32 + bkrow) * 64 + bnc0;
        _Float16 tmp[8];
        *(uint4*)tmp = *(const uint4*)bs;
#pragma unroll
        for (int j = 0; j < 8; ++j) lB[buf][(bnc0 + j) * 40 + bkrow] = tmp[j];
    };

    stageA(0, 0);
    stageB(0, 0);
    asm volatile("s_wait_asynccnt 0x0" ::: "memory");
    __syncthreads();

    for (int s = 0; s < 16; ++s) {
        const int cur = s & 1;
        if (s + 1 < 16) {
            stageA(s + 1, cur ^ 1);
            stageB(s + 1, cur ^ 1);
        }

        const int kh = (lane >> 4) * 8;
        const int kb = (lane >> 4) * 8;
#pragma unroll
        for (int mi = 0; mi < 2; ++mi) {
            const int rl = wm * 32 + mi * 16 + (lane & 15);
            v8u au;
#pragma unroll
            for (int v = 0; v < 8; ++v) {
                int kpos = ((v & 3) << 1) + ((v >= 4) ? 16 : 0) + kh;
                au[v] = *(const unsigned*)(&lA[cur][rl * 40 + kpos]);
            }
            v16h afr = __builtin_bit_cast(v16h, au);
#pragma unroll
            for (int ni = 0; ni < 2; ++ni) {
                const int ncol = wn * 32 + ni * 16 + (lane & 15);
                v8u bu;
#pragma unroll
                for (int v = 0; v < 8; ++v)
                    bu[v] = *(const unsigned*)(&lB[cur][ncol * 40 + kb + ((v & 3) << 1) + ((v >= 4) ? 16 : 0)]);
                v16h bfr = __builtin_bit_cast(v16h, bu);
                acc[mi][ni] = __builtin_amdgcn_wmma_f32_16x16x32_f16(
                    false, afr, false, bfr, (short)0, acc[mi][ni], false, false);
            }
        }

        if (s + 1 < 16) {
            asm volatile("s_wait_asynccnt 0x0" ::: "memory");
            __syncthreads();
        }
    }

#pragma unroll
    for (int mi = 0; mi < 2; ++mi)
#pragma unroll
        for (int ni = 0; ni < 2; ++ni) {
            const int nc = wn * 32 + ni * 16 + (lane & 15);
            float bias = 0.f;
            if (nc < 18)      bias = bcls[nc];
            else if (nc < 54) bias = bbx[nc - 18];
            const int mbase = m0 + wm * 32 + mi * 16 + ((lane >> 4) << 3);
#pragma unroll
            for (int r = 0; r < 8; ++r)
                head[(size_t)(mbase + r) * 64 + nc] = acc[mi][ni][r] + bias;
        }
}

// ---------------------------------------------------------------- anchors -> proposals/scores
__global__ __launch_bounds__(256) void k_proposals(const float* __restrict__ head,
                                                   const float* __restrict__ anchors,
                                                   const int* __restrict__ im,
                                                   float* __restrict__ props,
                                                   float* __restrict__ scores,
                                                   float* __restrict__ live) {
    int i = blockIdx.x * 256 + threadIdx.x;
    if (i >= NANCH) return;
    int m = i / NA, a = i % NA;
    const float* h = head + (size_t)m * 64;
    float score = 1.f / (1.f + expf(h[a] - h[NA + a]));   // sigmoid(pos-neg)
    float d0 = h[18 + 4 * a], d1 = h[19 + 4 * a], d2 = h[20 + 4 * a], d3 = h[21 + 4 * a];
    float ax1 = anchors[i * 4 + 0], ay1 = anchors[i * 4 + 1];
    float ax2 = anchors[i * 4 + 2], ay2 = anchors[i * 4 + 3];
    float w = ax2 - ax1 + 1.f, hh = ay2 - ay1 + 1.f;
    float cx = ax1 + 0.5f * w, cy = ay1 + 0.5f * hh;
    float pcx = d0 * w + cx, pcy = d1 * hh + cy;
    float pw = expf(d2) * w, ph = expf(d3) * hh;
    float lim = (float)(im[0] - 1);
    float x1 = fminf(fmaxf(pcx - 0.5f * pw, 0.f), lim);
    float y1 = fminf(fmaxf(pcy - 0.5f * ph, 0.f), lim);
    float x2 = fminf(fmaxf(pcx + 0.5f * pw, 0.f), lim);
    float y2 = fminf(fmaxf(pcy + 0.5f * ph, 0.f), lim);
    props[i * 4 + 0] = x1; props[i * 4 + 1] = y1;
    props[i * 4 + 2] = x2; props[i * 4 + 3] = y2;
    scores[i] = score;
    live[i]   = score;
}

// ---------------------------------------------------------------- greedy NMS (single WGP)
// wave32 shuffle argmax: 3 barriers per serial iteration instead of ~12
__global__ __launch_bounds__(1024) void k_nms(const float* __restrict__ boxes,
                                              float* __restrict__ live,
                                              int* __restrict__ idxbuf) {
    __shared__ float svw[32];
    __shared__ int   siw[32];
    __shared__ float sb[5];
    __shared__ int   scur;
    const int tid  = threadIdx.x;
    const int lane = tid & 31;
    const int wid  = tid >> 5;

    for (int it = 0; it < MAXOUT; ++it) {
        float best = -INFINITY; int bi = -1;
        for (int j = tid; j < NANCH; j += 1024) {
            float v = live[j];
            if (v > best) { best = v; bi = j; }
        }
#pragma unroll
        for (int o = 16; o > 0; o >>= 1) {           // intra-wave argmax, no barriers
            float ov = __shfl_xor(best, o, 32);
            int   oi = __shfl_xor(bi, o, 32);
            if (ov > best) { best = ov; bi = oi; }
        }
        if (lane == 0) { svw[wid] = best; siw[wid] = bi; }
        __syncthreads();
        if (wid == 0) {                               // wave 0 reduces 32 partials
            best = svw[lane]; bi = siw[lane];
#pragma unroll
            for (int o = 16; o > 0; o >>= 1) {
                float ov = __shfl_xor(best, o, 32);
                int   oi = __shfl_xor(bi, o, 32);
                if (ov > best) { best = ov; bi = oi; }
            }
            if (lane == 0) {
                idxbuf[it] = (best > -INFINITY) ? bi : -1;
                scur = bi;
                if (bi >= 0) {
                    float x1 = boxes[bi * 4], y1 = boxes[bi * 4 + 1];
                    float x2 = boxes[bi * 4 + 2], y2 = boxes[bi * 4 + 3];
                    sb[0] = x1; sb[1] = y1; sb[2] = x2; sb[3] = y2;
                    sb[4] = (x2 - x1) * (y2 - y1);
                }
            }
        }
        __syncthreads();
        int i = scur;
        if (i >= 0) {
            float bx1 = sb[0], by1 = sb[1], bx2 = sb[2], by2 = sb[3], ba = sb[4];
            for (int j = tid; j < NANCH; j += 1024) {
                float x1 = boxes[j * 4], y1 = boxes[j * 4 + 1];
                float x2 = boxes[j * 4 + 2], y2 = boxes[j * 4 + 3];
                float xx1 = fmaxf(x1, bx1), yy1 = fmaxf(y1, by1);
                float xx2 = fminf(x2, bx2), yy2 = fminf(y2, by2);
                float inter = fmaxf(xx2 - xx1, 0.f) * fmaxf(yy2 - yy1, 0.f);
                float area = (x2 - x1) * (y2 - y1);
                float iou = inter / (area + ba - inter + 1e-8f);
                if (iou > NMS_T || j == i) live[j] = -INFINITY;
            }
        }
        // suppression writes only need to be visible before next iteration's scan;
        // scan happens after the next __syncthreads() at top of loop body path above
        __syncthreads();
    }
}

// ---------------------------------------------------------------- gather outputs
__global__ __launch_bounds__(256) void k_gather(const int* __restrict__ idxbuf,
                                                const float* __restrict__ props,
                                                const float* __restrict__ scores,
                                                float* __restrict__ out) {
    int k = blockIdx.x * 256 + threadIdx.x;
    if (k >= MAXOUT) return;
    int i = idxbuf[k];
    float* roi = out + (size_t)k * 5;
    roi[0] = 0.f;
    if (i >= 0) {
        roi[1] = props[i * 4 + 0]; roi[2] = props[i * 4 + 1];
        roi[3] = props[i * 4 + 2]; roi[4] = props[i * 4 + 3];
        out[5 * MAXOUT + k] = scores[i];
    } else {
        roi[1] = roi[2] = roi[3] = roi[4] = -1.f;
        out[5 * MAXOUT + k] = -1.f;
    }
}

// ---------------------------------------------------------------- host launcher
extern "C" void kernel_launch(void* const* d_in, const int* in_sizes, int n_in,
                              void* d_out, int out_size, void* d_ws, size_t ws_size,
                              hipStream_t stream) {
    const float* feature = (const float*)d_in[0];
    const float* anchors = (const float*)d_in[1];
    const float* W_rpn   = (const float*)d_in[2];
    const float* b_rpn   = (const float*)d_in[3];
    const float* W_cls   = (const float*)d_in[4];
    const float* b_cls   = (const float*)d_in[5];
    const float* W_box   = (const float*)d_in[6];
    const float* b_box   = (const float*)d_in[7];
    const int*   im_size = (const int*)d_in[8];
    float* out = (float*)d_out;

    char* ws = (char*)d_ws;
    size_t off = 0;
    _Float16* wsWrpn = (_Float16*)(ws + off); off += (size_t)K9 * CCH * 2;          // 4.72 MB
    _Float16* wsPad  = (_Float16*)(ws + off); off += (size_t)98 * 98 * 512 * 2;     // 9.83 MB
    _Float16* wsX    = (_Float16*)(ws + off); off += (size_t)MROWS * CCH * 2;       // 9.44 MB
    _Float16* wsWcb  = (_Float16*)(ws + off); off += (size_t)512 * 64 * 2;
    float*    wsHead = (float*)(ws + off);    off += (size_t)MROWS * 64 * 4;        // 2.36 MB
    float*    wsProp = (float*)(ws + off);    off += (size_t)NANCH * 4 * 4;         // 1.33 MB
    float*    wsScr  = (float*)(ws + off);    off += (size_t)NANCH * 4;
    float*    wsLive = (float*)(ws + off);    off += (size_t)NANCH * 4;
    int*      wsIdx  = (int*)(ws + off);      off += (size_t)MAXOUT * 4;

    // 1) precision conversion + packing
    {
        int n = K9 * CCH;
        k_cvt_f16<<<(n + 255) / 256, 256, 0, stream>>>(W_rpn, wsWrpn, n);
    }
    {
        int n = 98 * 98 * 512;
        k_pad_feat<<<(n + 255) / 256, 256, 0, stream>>>(feature, wsPad);
    }
    k_pack_wcb<<<(512 * 64 + 255) / 256, 256, 0, stream>>>(W_cls, W_box, wsWcb);

    // 2) 3x3 conv as WMMA GEMM (M=9216, N=512, K=4608), async-LDS pipelined
    k_conv_gemm<<<dim3(CCH / 64, MROWS / 128), 256, 0, stream>>>(wsPad, wsWrpn, b_rpn, wsX);

    // 3) fused cls+box heads (M=9216, N=64, K=512)
    k_head_gemm<<<dim3(1, MROWS / 128), 256, 0, stream>>>(wsX, wsWcb, b_cls, b_box, wsHead);

    // 4) sigmoid scores + bbox transform + clip
    k_proposals<<<(NANCH + 255) / 256, 256, 0, stream>>>(wsHead, anchors, im_size,
                                                         wsProp, wsScr, wsLive);

    // 5) greedy NMS, 1000 serial iterations on one workgroup
    k_nms<<<1, 1024, 0, stream>>>(wsProp, wsLive, wsIdx);

    // 6) gather rois + scores into d_out (rois[1000,5] ++ scores[1000])
    k_gather<<<(MAXOUT + 255) / 256, 256, 0, stream>>>(wsIdx, wsProp, wsScr, out);
}
